// VanillaRNN_54623394071222
// MI455X (gfx1250) — compile-verified, matrix operand order checked by reading, not run on hardware
//
#include <hip/hip_runtime.h>
#include <hip/hip_bf16.h>

typedef __attribute__((ext_vector_type(16))) __bf16 v16bf;
typedef __attribute__((ext_vector_type(8)))  float  v8f;

#define Hdim 512
#define Bdim 512
#define Sdim 1024
#define Odim 10
#define BM   16     // batch rows per block
#define KT   32     // K per WMMA
// W fragment layout: block fb = kt*32 + jt holds the 32x16 (KxN) bf16 tile,
// 512 elements, lane l owns elements [l*16, l*16+16):
//   k = kt*32 + (l>>4)*16 + i,  n = jt*16 + (l&15)

// gfx1250 hardware tanh (TRANS op). Two v_nops cover the documented TRANS
// result-use hazard since the compiler can't schedule around the asm block.
__device__ __forceinline__ float htanh(float v) {
    float r;
    asm("v_tanh_f32 %0, %1\n\tv_nop\n\tv_nop" : "=v"(r) : "v"(v));
    return r;
}

__global__ void swizzle_whh(const float* __restrict__ Whh, __bf16* __restrict__ Wf) {
    int idx  = blockIdx.x * blockDim.x + threadIdx.x;   // 0 .. 512*512-1
    int i    = idx & 15;
    int lane = (idx >> 4) & 31;
    int jt   = (idx >> 9) & 31;
    int kt   = idx >> 14;
    int k = kt * 32 + ((lane >> 4) << 4) + i;
    int j = jt * 16 + (lane & 15);
    // B(k, n) = W_hh[n, k]  (since we need h @ W_hh^T)
    Wf[idx] = (__bf16)Whh[j * Hdim + k];
}

__global__ void __launch_bounds__(512)
rnn_persistent(const float* __restrict__ x,   const float* __restrict__ Whx,
               const float* __restrict__ bhx, const float* __restrict__ bhh,
               const __bf16* __restrict__ Wf, float* __restrict__ hfin)
{
    __shared__ __bf16 h_lds[2][BM][Hdim];   // 32 KB, double-buffered state (bf16)
    __shared__ float  x_t[Sdim][BM];        // 64 KB, x rows TRANSPOSED: x_t[s][m]
    __shared__ float  whx_lds[Hdim];        // 2 KB
    __shared__ float  bias_lds[Hdim];       // 2 KB  (b_hx + b_hh fused)

    const int tid  = threadIdx.x;
    const int lane = tid & 31;
    const int wave = tid >> 5;              // 16 waves
    const int row0 = blockIdx.x * BM;

    // coalesced global read, transposed LDS write
    for (int i = tid; i < BM * Sdim; i += 512) {
        int r = i >> 10, c = i & (Sdim - 1);
        x_t[c][r] = x[(row0 + r) * Sdim + c];
    }
    for (int i = tid; i < Hdim; i += 512) {
        whx_lds[i]  = Whx[i];
        bias_lds[i] = bhx[i] + bhh[i];
    }
    for (int i = tid; i < BM * Hdim; i += 512)
        h_lds[0][i >> 9][i & (Hdim - 1)] = (__bf16)0.0f;   // h0 = 0
    __syncthreads();

    const int mrow = lane & 15;             // A-fragment row
    const int ksub = (lane >> 4) << 4;      // A-fragment K sub-offset (0 / 16)
    const int jt0  = wave * 2;              // this wave's two N tiles
    const int jt1  = wave * 2 + 1;
    const int crow = (lane >> 4) * 8;       // C/D layout: lanes 16-31 -> M+8
    const int col0 = jt0 * 16 + (lane & 15);
    const int col1 = jt1 * 16 + (lane & 15);
    const float xw0 = whx_lds[col0], bb0 = bias_lds[col0];
    const float xw1 = whx_lds[col1], bb1 = bias_lds[col1];

    int cur = 0;
    for (int s = 0; s < Sdim; ++s) {
        v8f acc0 = {}, acc1 = {};
        #pragma unroll 4
        for (int kt = 0; kt < Hdim / KT; ++kt) {
            // prefetch next K block of W (wraps: primes kt=0 for next step)
            int kn = (kt + 1) & 15;
            __builtin_prefetch(&Wf[(((kn * 32 + jt0) << 9)) + lane * 16], 0, 1);

            const v16bf a  = *(const v16bf*)&h_lds[cur][mrow][kt * KT + ksub];
            const v16bf b0 = *(const v16bf*)&Wf[((kt * 32 + jt0) << 9) + lane * 16];
            const v16bf b1 = *(const v16bf*)&Wf[((kt * 32 + jt1) << 9) + lane * 16];
            acc0 = __builtin_amdgcn_wmma_f32_16x16x32_bf16(false, a, false, b0,
                                                           (short)0, acc0, false, false);
            acc1 = __builtin_amdgcn_wmma_f32_16x16x32_bf16(false, a, false, b1,
                                                           (short)0, acc1, false, false);
        }
        const int nxt = cur ^ 1;
        // 8 contiguous x values for this lane's C rows: 2x ds_load_b128
        const v8f xv = *(const v8f*)&x_t[s][crow];
        #pragma unroll
        for (int r = 0; r < 8; ++r) {
            int m = r + crow;
            float h0 = htanh(fmaf(xv[r], xw0, bb0) + acc0[r]);
            float h1 = htanh(fmaf(xv[r], xw1, bb1) + acc1[r]);
            h_lds[nxt][m][col0] = (__bf16)h0;
            h_lds[nxt][m][col1] = (__bf16)h1;
        }
        __syncthreads();    // one barrier per step: writes done before next reads
        cur = nxt;
    }

    for (int i = tid; i < BM * Hdim; i += 512) {
        int r = i >> 9, c = i & (Hdim - 1);
        hfin[(row0 + r) * Hdim + c] = (float)h_lds[cur][r][c];
    }
}

__global__ void out_kernel(const float* __restrict__ hfin, const float* __restrict__ Woh,
                           const float* __restrict__ boh, float* __restrict__ out)
{
    int b = blockIdx.x;
    int lane = threadIdx.x;       // 32 threads (one wave)
    float p[Odim];
    #pragma unroll
    for (int j = 0; j < Odim; ++j) p[j] = 0.f;
    for (int k = lane; k < Hdim; k += 32) {
        float hv = hfin[b * Hdim + k];
        #pragma unroll
        for (int j = 0; j < Odim; ++j) p[j] += hv * Woh[j * Hdim + k];
    }
    #pragma unroll
    for (int j = 0; j < Odim; ++j) {
        #pragma unroll
        for (int off = 16; off > 0; off >>= 1)
            p[j] += __shfl_xor(p[j], off, 32);
    }
    if (lane == 0) {
        float m = -1e30f;
        #pragma unroll
        for (int j = 0; j < Odim; ++j) { p[j] += boh[j]; m = fmaxf(m, p[j]); }
        float sum = 0.f;
        #pragma unroll
        for (int j = 0; j < Odim; ++j) { p[j] = __expf(p[j] - m); sum += p[j]; }
        float inv = 1.f / sum;
        #pragma unroll
        for (int j = 0; j < Odim; ++j) out[b * Odim + j] = p[j] * inv;
    }
}

extern "C" void kernel_launch(void* const* d_in, const int* in_sizes, int n_in,
                              void* d_out, int out_size, void* d_ws, size_t ws_size,
                              hipStream_t stream) {
    const float* x   = (const float*)d_in[0];   // [512, 1024]
    const float* Whx = (const float*)d_in[1];   // [512, 1] -> 512
    const float* bhx = (const float*)d_in[2];   // [512]
    const float* Whh = (const float*)d_in[3];   // [512, 512]
    const float* bhh = (const float*)d_in[4];   // [512]
    const float* Woh = (const float*)d_in[5];   // [10, 512]
    const float* boh = (const float*)d_in[6];   // [10]
    float* out = (float*)d_out;                 // [512, 10] f32

    __bf16* Wf  = (__bf16*)d_ws;                                        // 512 KB
    float* hfin = (float*)((char*)d_ws + Hdim * Hdim * sizeof(__bf16)); // 1 MB

    swizzle_whh<<<(Hdim * Hdim) / 256, 256, 0, stream>>>(Whh, Wf);
    rnn_persistent<<<Bdim / BM, 512, 0, stream>>>(x, Whx, bhx, bhh, Wf, hfin);
    out_kernel<<<Bdim, 32, 0, stream>>>(hfin, Woh, boh, out);
}
